// OriginalFullAttention_87780541596007
// MI455X (gfx1250) — matrix operand order repaired and checked by reading, not compiled, not run on hardware
//
#include <hip/hip_runtime.h>

typedef __bf16 bf16_t;
typedef __attribute__((ext_vector_type(16))) __bf16 v16bf;
typedef __attribute__((ext_vector_type(8)))  float  v8f;
typedef __attribute__((ext_vector_type(4)))  int    v4i;

#define BATCH     2
#define NHEAD     16
#define NHEADS_T  (BATCH * NHEAD)   // 32
#define SEQ       2048
#define HDIM      64
#define KV_STEP   64
#define NSTEPS    (SEQ / KV_STEP)   // 32
#define K_STRIDE  72   // halfs per K-tile row   (64 + 8 pad)
#define VT_STRIDE 72   // halfs per V^T row      (64 + 8 pad)
#define P_STRIDE  72   // halfs per P row        (64 + 8 pad)
#define TILE_ELEM (KV_STEP * HDIM)               // 4096 halfs per tile
#define WS_TENSOR ((size_t)NHEADS_T * SEQ * HDIM)     // 4194304 bf16 elems
#define WS_NEEDED (WS_TENSOR * 2 * 2 + (size_t)BATCH * SEQ * 4)  // 16793600 B

#if defined(__has_builtin)
#if __has_builtin(__builtin_amdgcn_global_load_async_to_lds_b128)
#define HAVE_ASYNC_LDS 1
#endif
#endif

union V16U { v16bf v; uint4 q[2]; };

static __device__ __forceinline__ float red_max16(float v) {
#pragma unroll
  for (int m = 1; m < 16; m <<= 1) v = fmaxf(v, __shfl_xor(v, m, 16));
  return v;
}

static __device__ __forceinline__ v8f wmma_bf16(v16bf a, v16bf bm, v8f c) {
  return __builtin_amdgcn_wmma_f32_16x16x32_bf16(false, a, false, bm,
                                                 (short)0, c, false, false);
}

static __device__ __forceinline__ void wait_stage_async() {
#if defined(HAVE_ASYNC_LDS)
#if __has_builtin(__builtin_amdgcn_s_wait_asynccnt)
  __builtin_amdgcn_s_wait_asynccnt(0);
#else
  asm volatile("s_wait_asynccnt 0x0" ::: "memory");
#endif
#endif
}

// ---------- pre-pass: f32 K/V -> bf16 workspace (K tiled row-major,
// V tiled transposed), plus precomputed additive mask ----------
__global__ __launch_bounds__(256) void prep_bf16(
    const float* __restrict__ K, const float* __restrict__ V,
    const float* __restrict__ Msk, bf16_t* __restrict__ Kws,
    bf16_t* __restrict__ Vws, float* __restrict__ MA) {
  const int gid = blockIdx.x * 256 + threadIdx.x;   // one float4 of K and V
  const size_t off = (size_t)gid * 4;               // element offset
  const int head = (int)(off >> 17);                // / (SEQ*HDIM)
  const int rem  = (int)(off & (SEQ * HDIM - 1));
  const int key  = rem >> 6;
  const int h    = rem & 63;
  const int tile = key >> 6;
  const int kl   = key & 63;
  const float4 kq = *(const float4*)(K + off);
  const float4 vq = *(const float4*)(V + off);
  const size_t tb = (size_t)(head * NSTEPS + tile) * TILE_ELEM;
  bf16_t* kd = Kws + tb + (size_t)kl * HDIM + h;
  kd[0] = (bf16_t)kq.x; kd[1] = (bf16_t)kq.y;
  kd[2] = (bf16_t)kq.z; kd[3] = (bf16_t)kq.w;
  bf16_t* vd = Vws + tb + (size_t)h * HDIM + kl;    // [h][key] within tile
  vd[0 * HDIM] = (bf16_t)vq.x; vd[1 * HDIM] = (bf16_t)vq.y;
  vd[2 * HDIM] = (bf16_t)vq.z; vd[3 * HDIM] = (bf16_t)vq.w;
  if (gid < BATCH * SEQ) MA[gid] = (1.0f - Msk[gid]) * -10000.0f;
}

// ---------- staging helpers ----------
// prestaged bf16 tiles (contiguous 64x64 halfs) -> padded LDS
static __device__ __forceinline__ void stage_pre(const bf16_t* __restrict__ kt,
                                                 const bf16_t* __restrict__ vt,
                                                 bf16_t* kb, bf16_t* vb, int tid) {
#pragma unroll
  for (int it = 0; it < 4; ++it) {
    const int idx = tid + it * 128;   // 0..511 uint4s
    const int row = idx >> 3;
    const int c   = idx & 7;
#if defined(HAVE_ASYNC_LDS)
    __builtin_amdgcn_global_load_async_to_lds_b128(
        (v4i*)(kt + (size_t)idx * 8),
        (v4i*)(kb + row * K_STRIDE + c * 8), 0, 0);
    __builtin_amdgcn_global_load_async_to_lds_b128(
        (v4i*)(vt + (size_t)idx * 8),
        (v4i*)(vb + row * VT_STRIDE + c * 8), 0, 0);
#else
    *(uint4*)(kb + row * K_STRIDE + c * 8) = *(const uint4*)(kt + (size_t)idx * 8);
    *(uint4*)(vb + row * VT_STRIDE + c * 8) = *(const uint4*)(vt + (size_t)idx * 8);
#endif
  }
}

// fallback: stage straight from f32 K/V with on-the-fly convert + transpose
static __device__ __forceinline__ void stage_f32(const float* __restrict__ Kg,
                                                 const float* __restrict__ Vg,
                                                 bf16_t* kb, bf16_t* vb, int tid) {
#pragma unroll
  for (int it = 0; it < 8; ++it) {
    const int idx = tid + it * 128;
    const int row = idx >> 4;
    const int h   = (idx & 15) * 4;
    const float4 kq = *(const float4*)(Kg + row * HDIM + h);
    const float4 vq = *(const float4*)(Vg + row * HDIM + h);
    kb[row * K_STRIDE + h + 0] = (bf16_t)kq.x;
    kb[row * K_STRIDE + h + 1] = (bf16_t)kq.y;
    kb[row * K_STRIDE + h + 2] = (bf16_t)kq.z;
    kb[row * K_STRIDE + h + 3] = (bf16_t)kq.w;
    vb[(h + 0) * VT_STRIDE + row] = (bf16_t)vq.x;
    vb[(h + 1) * VT_STRIDE + row] = (bf16_t)vq.y;
    vb[(h + 2) * VT_STRIDE + row] = (bf16_t)vq.z;
    vb[(h + 3) * VT_STRIDE + row] = (bf16_t)vq.w;
  }
}

// ---------- main flash-attention kernel ----------
// grid.x = BATCH*NHEAD, grid.y = SEQ/64 ; block = 128 (4 wave32s)
template <bool PRE>
__global__ __launch_bounds__(128) void flash_attn_bf16_wmma(
    const float* __restrict__ Q, const float* __restrict__ K,
    const float* __restrict__ V, const float* __restrict__ Msk,
    const float* __restrict__ MA, const bf16_t* __restrict__ Kws,
    const bf16_t* __restrict__ Vws, float* __restrict__ Out) {
  __shared__ __align__(16) bf16_t lds_k[2][KV_STEP * K_STRIDE];
  __shared__ __align__(16) bf16_t lds_vt[2][HDIM * VT_STRIDE];
  __shared__ __align__(16) bf16_t lds_p[4 * 16 * P_STRIDE];

  const int head = blockIdx.x;
  const int b    = head >> 4;
  const int n    = head & 15;
  const int tid  = threadIdx.x;
  const int wave = tid >> 5;
  const int lane = tid & 31;
  const int hh   = lane >> 4;
  const int lm   = lane & 15;
  const int q0   = blockIdx.y * 64 + wave * 16;

  const size_t headBase = (size_t)head * SEQ * HDIM;
  const float* Kh = K + headBase;
  const float* Vh = V + headBase;
  const bf16_t* Kt = Kws + (size_t)head * NSTEPS * TILE_ELEM;
  const bf16_t* Vt = Vws + (size_t)head * NSTEPS * TILE_ELEM;

  // Q A-fragments (16x32 bf16), chunk c covers h in [c*32, c*32+32)
  v16bf aq[2];
  {
    const float* qb = Q + headBase + (size_t)(q0 + lm) * HDIM;
#pragma unroll
    for (int c = 0; c < 2; ++c) {
#pragma unroll
      for (int e = 0; e < 8; ++e) {
        aq[c][e]     = (bf16_t)qb[c * 32 + 16 * hh + e];
        aq[c][8 + e] = (bf16_t)qb[c * 32 + 16 * hh + 8 + e];
      }
    }
  }

  v16bf ones;
#pragma unroll
  for (int e = 0; e < 16; ++e) ones[e] = (bf16_t)1.0f;

  v8f o[4] = {};
  float m_run[8], l_run[8];
#pragma unroll
  for (int r = 0; r < 8; ++r) { m_run[r] = -1e30f; l_run[r] = 0.0f; }

  const float scale = 0.125f;

  if (PRE) stage_pre(Kt, Vt, lds_k[0], lds_vt[0], tid);
  else     stage_f32(Kh, Vh, lds_k[0], lds_vt[0], tid);
  if (PRE) wait_stage_async();
  __syncthreads();

  for (int step = 0; step < NSTEPS; ++step) {
    const int kv0 = step * KV_STEP;
    const int cur = step & 1;
    if (step + 1 < NSTEPS) {
      if (PRE)
        stage_pre(Kt + (size_t)(step + 1) * TILE_ELEM,
                  Vt + (size_t)(step + 1) * TILE_ELEM,
                  lds_k[cur ^ 1], lds_vt[cur ^ 1], tid);
      else
        stage_f32(Kh + (size_t)(kv0 + KV_STEP) * HDIM,
                  Vh + (size_t)(kv0 + KV_STEP) * HDIM,
                  lds_k[cur ^ 1], lds_vt[cur ^ 1], tid);
    }

    const bf16_t* kb = lds_k[cur];
    const bf16_t* vb = lds_vt[cur];

    // scores: 4 tiles of 16x16, contraction over H=64 (2 WMMAs each)
    v8f sc[4];
#pragma unroll
    for (int t = 0; t < 4; ++t) {
      const bf16_t* kp = kb + (16 * t + lm) * K_STRIDE;
      V16U b0, b1;
      b0.q[0] = *(const uint4*)(kp +      16 * hh);
      b0.q[1] = *(const uint4*)(kp +      16 * hh + 8);
      b1.q[0] = *(const uint4*)(kp + 32 + 16 * hh);
      b1.q[1] = *(const uint4*)(kp + 32 + 16 * hh + 8);
      v8f z = {};
      v8f acc = wmma_bf16(aq[1], b1.v, z);
      sc[t]   = wmma_bf16(aq[0], b0.v, acc);
      const int key = kv0 + 16 * t + lm;
      const float madd = PRE ? MA[b * SEQ + key]
                             : (1.0f - Msk[b * SEQ + key]) * -10000.0f;
#pragma unroll
      for (int e = 0; e < 8; ++e) sc[t][e] = sc[t][e] * scale + madd;
    }

    // online softmax (max via shfl tree; sums via WMMA below)
    float alpha_s[8];
#pragma unroll
    for (int r = 0; r < 8; ++r) {
      const float cmax =
          fmaxf(fmaxf(sc[0][r], sc[1][r]), fmaxf(sc[2][r], sc[3][r]));
      const float rmax = red_max16(cmax);
      const float mnew = fmaxf(m_run[r], rmax);
      const float a    = __expf(m_run[r] - mnew);
      m_run[r] = mnew;
      alpha_s[r] = a;
#pragma unroll
      for (int t = 0; t < 4; ++t) sc[t][r] = __expf(sc[t][r] - mnew);
#pragma unroll
      for (int j = 0; j < 4; ++j) o[j][r] *= a;
    }

    // P: C-layout -> LDS -> A-layout (wave-private, in-order DS)
    bf16_t* pw = lds_p + wave * (16 * P_STRIDE);
#pragma unroll
    for (int t = 0; t < 4; ++t)
#pragma unroll
      for (int r = 0; r < 8; ++r)
        pw[(r + 8 * hh) * P_STRIDE + 16 * t + lm] = (bf16_t)sc[t][r];

    V16U pa[2];
#pragma unroll
    for (int c = 0; c < 2; ++c) {
      const bf16_t* pr = pw + lm * P_STRIDE + c * 32;
      pa[c].q[0] = *(const uint4*)(pr + 16 * hh);
      pa[c].q[1] = *(const uint4*)(pr + 16 * hh + 8);
    }

    // row sums via WMMA against all-ones B
    {
      v8f z = {};
      v8f s1   = wmma_bf16(pa[1].v, ones, z);
      v8f sums = wmma_bf16(pa[0].v, ones, s1);
#pragma unroll
      for (int r = 0; r < 8; ++r)
        l_run[r] = l_run[r] * alpha_s[r] + sums[r];
    }

    // O += P (16x64) x V (64x64)
#pragma unroll
    for (int j = 0; j < 4; ++j) {
      const bf16_t* vp = vb + (j * 16 + lm) * VT_STRIDE + 16 * hh;
      V16U v0, v1;
      v0.q[0] = *(const uint4*)(vp);
      v0.q[1] = *(const uint4*)(vp + 8);
      v1.q[0] = *(const uint4*)(vp + 32);
      v1.q[1] = *(const uint4*)(vp + 32 + 8);
      v8f tmp = wmma_bf16(pa[0].v, v0.v, o[j]);
      o[j]    = wmma_bf16(pa[1].v, v1.v, tmp);
    }

    if (PRE) wait_stage_async();
    __syncthreads();
  }

  // epilogue: normalize and store Out[b][f][n][h]
#pragma unroll
  for (int r = 0; r < 8; ++r) {
    const int row = r + 8 * hh;
    const float inv = 1.0f / l_run[r];
    const size_t obase =
        (((size_t)b * SEQ + (size_t)(q0 + row)) * NHEAD + n) * HDIM;
#pragma unroll
    for (int j = 0; j < 4; ++j)
      Out[obase + j * 16 + lm] = o[j][r] * inv;
  }
}

extern "C" void kernel_launch(void* const* d_in, const int* in_sizes, int n_in,
                              void* d_out, int out_size, void* d_ws, size_t ws_size,
                              hipStream_t stream) {
  const float* Q = (const float*)d_in[0];
  const float* K = (const float*)d_in[1];
  const float* V = (const float*)d_in[2];
  const float* M = (const float*)d_in[3];
  (void)in_sizes; (void)n_in; (void)out_size;
  dim3 grid(NHEADS_T, SEQ / 64);
  dim3 block(128);
  if (ws_size >= WS_NEEDED) {
    bf16_t* Kws = (bf16_t*)d_ws;
    bf16_t* Vws = Kws + WS_TENSOR;
    float*  MA  = (float*)(Vws + WS_TENSOR);
    const int n4 = (int)(WS_TENSOR / 4);          // float4s per tensor
    prep_bf16<<<n4 / 256, 256, 0, stream>>>(K, V, M, Kws, Vws, MA);
    flash_attn_bf16_wmma<true><<<grid, block, 0, stream>>>(
        Q, K, V, M, MA, Kws, Vws, (float*)d_out);
  } else {
    flash_attn_bf16_wmma<false><<<grid, block, 0, stream>>>(
        Q, K, V, M, nullptr, nullptr, nullptr, (float*)d_out);
  }
}
